// AttentionBlock_28054726378259
// MI455X (gfx1250) — matrix operand order repaired and checked by reading, not compiled, not run on hardware
//
#include <hip/hip_runtime.h>

// ---------------------------------------------------------------------------
// AttentionBlock for MI455X (gfx1250, wave32, WMMA + async-to-LDS).
// B=4, H=W=64, C=256, N=H*W=4096, groups=8.
//
// groupnorm -> f16 QKV WMMA-GEMM -> flash attention (async K staging, double
// buffered; register-pipelined V; WMMA QK^T and PV with online softmax; the
// 4096x4096 score matrix never touches HBM) -> WMMA proj GEMM + residual.
//
// Workspace layout (bytes), total ~59.3 MB:
//   xn   f32 [16384][256]      @ 0          (16777216)
//   xh   f16 [16384][256]      @ 16777216   (8388608)
//   wh   f16 4x[256][256]      @ 25165824   (524288)
//   q    f16 [16384][256]      @ 25690112   (8388608)
//   k    f16 [16384][256]      @ 34078720   (8388608)
//   v    f16 [16384][256]      @ 42467328   (8388608)
//   ao   f16 [16384][256]      @ 50855936   (8388608)
//   stats f32 [32][2]          @ 59244544   (256)
// ---------------------------------------------------------------------------

typedef unsigned short u16;
typedef __attribute__((ext_vector_type(16))) _Float16 v16h;
typedef __attribute__((ext_vector_type(8)))  float    v8f;

union Frag16 {      // one WMMA 16-bit A/B fragment = 16 halves = 2x b128
  v16h v;
  uint4 q[2];
};

static __device__ __forceinline__ u16 f2h(float f) {
  _Float16 h = (_Float16)f;
  return __builtin_bit_cast(unsigned short, h);
}

static __device__ __forceinline__ v8f wmma_f16(v16h a, v16h b, v8f c) {
  // D = A(16x32 f16) * B(32x16 f16) + C(16x16 f32)
  return __builtin_amdgcn_wmma_f32_16x16x32_f16(
      /*neg_a=*/false, a, /*neg_b=*/false, b,
      /*c_mod=*/(short)0, c, /*reuse_a=*/false, /*reuse_b=*/false);
}

// CDNA5 async DMA: global -> LDS, 16B per lane, tracked by ASYNCcnt.
// GVS addressing: mem = SADDR(64b, SGPR) + VADDR(32b, VGPR) + IOFFSET.
// First operand is the per-lane LDS byte address (low 32 bits of the generic
// pointer to __shared__ are the LDS offset on gfx1250).
static __device__ __forceinline__ void async_ld_b128(
    unsigned lds_addr, unsigned goff, unsigned long long base) {
  asm volatile("global_load_async_to_lds_b128 %0, %1, %2"
               :: "v"(lds_addr), "v"(goff), "s"(base) : "memory");
}

#define NTOK 16384   // B*H*W
#define CCH  256
#define NPB  4096    // tokens per batch

// --------------------------- GroupNorm ------------------------------------
__global__ __launch_bounds__(256) void gn_stats_kernel(
    const float* __restrict__ x, float* __restrict__ stats) {
  int bg = blockIdx.x;                 // 32 blocks: b*8+g
  int b = bg >> 3, g = bg & 7;
  const float* xb = x + (size_t)b * NPB * CCH + g * 32;
  __shared__ float ssum[256], ssq[256];
  float s = 0.f, s2 = 0.f;
  for (int i = threadIdx.x; i < NPB * 8; i += 256) {
    int n = i >> 3, c4 = (i & 7) * 4;
    float4 vv = *(const float4*)(xb + (size_t)n * CCH + c4);
    s  += vv.x + vv.y + vv.z + vv.w;
    s2 += vv.x * vv.x + vv.y * vv.y + vv.z * vv.z + vv.w * vv.w;
  }
  ssum[threadIdx.x] = s; ssq[threadIdx.x] = s2;
  __syncthreads();
  for (int st = 128; st > 0; st >>= 1) {
    if ((int)threadIdx.x < st) {
      ssum[threadIdx.x] += ssum[threadIdx.x + st];
      ssq[threadIdx.x]  += ssq[threadIdx.x + st];
    }
    __syncthreads();
  }
  if (threadIdx.x == 0) {
    const float invn = 1.f / 131072.f;
    float mean = ssum[0] * invn;
    float var  = ssq[0] * invn - mean * mean;
    stats[bg * 2 + 0] = mean;
    stats[bg * 2 + 1] = rsqrtf(var + 1e-3f);
  }
}

__global__ __launch_bounds__(256) void gn_apply_kernel(
    const float* __restrict__ x, const float* __restrict__ gamma,
    const float* __restrict__ beta, const float* __restrict__ stats,
    float* __restrict__ xn, u16* __restrict__ xh) {
  size_t i = ((size_t)blockIdx.x * 256 + threadIdx.x) * 4;  // 4096 blocks
  int c = (int)(i & 255);
  size_t n = i >> 8;
  int b = (int)(n >> 12);
  int g = c >> 5;
  const float* st = stats + (b * 8 + g) * 2;
  float mean = st[0], rstd = st[1];
  float4 vv = *(const float4*)(x + i);
  float4 gm = *(const float4*)(gamma + c);
  float4 bt = *(const float4*)(beta + c);
  float4 r;
  r.x = (vv.x - mean) * rstd * gm.x + bt.x;
  r.y = (vv.y - mean) * rstd * gm.y + bt.y;
  r.z = (vv.z - mean) * rstd * gm.z + bt.z;
  r.w = (vv.w - mean) * rstd * gm.w + bt.w;
  *(float4*)(xn + i) = r;
  ushort4 h;
  h.x = f2h(r.x); h.y = f2h(r.y); h.z = f2h(r.z); h.w = f2h(r.w);
  *(ushort4*)(xh + i) = h;
}

// --------------------------- weight f32 -> f16 -----------------------------
__global__ __launch_bounds__(256) void w2h_kernel(
    const float* __restrict__ w, u16* __restrict__ wh, int n) {
  int i = blockIdx.x * 256 + threadIdx.x;
  if (i < n) wh[i] = f2h(w[i]);
}

// --------------------------- QKV WMMA GEMM --------------------------------
#define LDA 40  // As row stride (32 k + 8 pad) halves
#define LDB 40  // Bs row stride (col-major k)  halves
__global__ __launch_bounds__(128) void qkv_gemm_kernel(
    const u16* __restrict__ Xh, const u16* __restrict__ Wh,
    const float* __restrict__ bq, const float* __restrict__ bk,
    const float* __restrict__ bv,
    u16* __restrict__ qo, u16* __restrict__ ko, u16* __restrict__ vo) {
  __shared__ __align__(16) u16 As[64 * LDA];
  __shared__ __align__(16) u16 Bs[64 * LDB];  // transposed: [col][k]
  const int tid = threadIdx.x, wave = tid >> 5, lane = tid & 31;
  const int hi = lane >> 4, ln = lane & 15;
  const int mbase = blockIdx.x * 64, nbase = blockIdx.y * 64;
  const int z = blockIdx.z;
  const u16* W = Wh + (size_t)z * 65536;
  const float* bias = (z == 0) ? bq : ((z == 1) ? bk : bv);
  u16* out = (z == 0) ? qo : ((z == 1) ? ko : vo);
  const float alpha = (z == 0) ? 0.0625f : 1.0f;  // 256^-0.5 folded into q

  const v8f z8 = {0.f, 0.f, 0.f, 0.f, 0.f, 0.f, 0.f, 0.f};
  v8f acc[4] = {z8, z8, z8, z8};

  for (int kb = 0; kb < CCH; kb += 32) {
    {  // stage A tile 64x32 (row major, b128 in / b128 out)
      int r = tid >> 1, cs = (tid & 1) * 16;
      const uint4* g = (const uint4*)(Xh + (size_t)(mbase + r) * CCH + kb + cs);
      *(uint4*)(As + r * LDA + cs)     = g[0];
      *(uint4*)(As + r * LDA + cs + 8) = g[1];
    }
    {  // stage B tile 32x64 transposed: Bs[col][kk] = W[kb+kk][nbase+col]
      int col = tid >> 1, ks = (tid & 1) * 16;
      __align__(16) u16 tmp[16];
      for (int j = 0; j < 16; ++j)
        tmp[j] = W[(size_t)(kb + ks + j) * CCH + nbase + col];
      *(uint4*)(Bs + col * LDB + ks)     = *(uint4*)tmp;
      *(uint4*)(Bs + col * LDB + ks + 8) = *(uint4*)(tmp + 8);
    }
    __syncthreads();
    Frag16 a;  // A layout: lo lanes K 0-15, hi lanes K 16-31
    {
      int r = wave * 16 + ln, kh = hi * 16;
      a.q[0] = *(const uint4*)(As + r * LDA + kh);
      a.q[1] = *(const uint4*)(As + r * LDA + kh + 8);
    }
    for (int nt = 0; nt < 4; ++nt) {
      Frag16 b;  // B layout (same k order as A): lane=col, lo lanes K0-15, hi lanes K16-31
      int col = nt * 16 + ln, kk = hi * 16;
      b.q[0] = *(const uint4*)(Bs + col * LDB + kk);
      b.q[1] = *(const uint4*)(Bs + col * LDB + kk + 8);
      acc[nt] = wmma_f16(a.v, b.v, acc[nt]);
    }
    __syncthreads();
  }
  for (int nt = 0; nt < 4; ++nt) {
    int col = nbase + nt * 16 + ln;
    float bsv = bias[col];
    for (int j = 0; j < 8; ++j) {  // C layout: M = j + 8*hi, N = ln
      int row = mbase + wave * 16 + j + 8 * hi;
      out[(size_t)row * CCH + col] = f2h((acc[nt][j] + bsv) * alpha);
    }
  }
}

// --------------------------- Flash attention -------------------------------
// Block: 4 waves x 16 query rows; grid (4096/64, B).
// K chunks: double-buffered async DMA to LDS. V chunks: register-pipelined.
#define LDK 264  // K-chunk LDS row stride: 256 + 8 pad (halves)
#define LDV 40   // V^T LDS row stride: 32 keys + 8 pad
#define LDP 40   // P  LDS row stride
#define KBUF (32 * LDK)
__global__ __launch_bounds__(128) void flash_attn_kernel(
    const u16* __restrict__ Q, const u16* __restrict__ K,
    const u16* __restrict__ V, u16* __restrict__ O) {
  __shared__ __align__(16) u16 Kl[2 * KBUF];       // [buf][key][chan]
  __shared__ __align__(16) u16 Vt[CCH * LDV];      // [chan][key] (transposed)
  __shared__ __align__(16) u16 Pl[4 * 16 * LDP];   // per-wave P scratch
  const int tid = threadIdx.x, wave = tid >> 5, lane = tid & 31;
  const int hi = lane >> 4, ln = lane & 15;
  const size_t bOff = (size_t)blockIdx.y * NPB * CCH;
  const int qbase = blockIdx.x * 64 + wave * 16;

  // Per-thread staging geometry: 8 b128 packets per chunk (1024 total/block).
  int skey[8], sc8[8];
  unsigned klds[2][8];
  for (int it = 0; it < 8; ++it) {
    int idx = tid + it * 128;
    skey[it] = idx >> 5;
    sc8[it]  = (idx & 31) * 8;
    klds[0][it] = (unsigned)(uintptr_t)(Kl + skey[it] * LDK + sc8[it]);
    klds[1][it] = klds[0][it] + KBUF * 2 /*bytes*/;
  }

  // Q fragments for this wave's 16 rows, full C=256 (8 k-tiles), in VGPRs
  Frag16 qa[8];
  {
    const u16* qrow = Q + bOff + (size_t)(qbase + ln) * CCH;
    for (int kt = 0; kt < 8; ++kt) {
      int kh = kt * 32 + hi * 16;
      qa[kt].q[0] = *(const uint4*)(qrow + kh);
      qa[kt].q[1] = *(const uint4*)(qrow + kh + 8);
    }
  }
  const v8f z8 = {0.f, 0.f, 0.f, 0.f, 0.f, 0.f, 0.f, 0.f};
  v8f o[16];
  for (int nt = 0; nt < 16; ++nt) o[nt] = z8;
  float m[8], l[8];
  for (int j = 0; j < 8; ++j) { m[j] = -1e30f; l[j] = 0.f; }

  // Async K stage: 8 DMA packets per wave per chunk (in-order completion).
  auto issueK = [&](int kc, int buf) {
    unsigned long long base =
        (unsigned long long)(const void*)(K + bOff + (size_t)kc * CCH);
    for (int it = 0; it < 8; ++it) {
      unsigned goff = (unsigned)((skey[it] * CCH + sc8[it]) * 2);
      async_ld_b128(klds[buf][it], goff, base);
    }
  };
  auto prefV = [&](int kc, uint4* vr) {
    const u16* Vg = V + bOff + (size_t)kc * CCH;
    for (int it = 0; it < 8; ++it)
      vr[it] = *(const uint4*)(Vg + (size_t)skey[it] * CCH + sc8[it]);
  };
  auto storeV = [&](const uint4* vr) {
    for (int it = 0; it < 8; ++it) {
      const u16* u = (const u16*)&vr[it];
      for (int j = 0; j < 8; ++j) Vt[(sc8[it] + j) * LDV + skey[it]] = u[j];
    }
  };

  uint4 vreg[8];
  prefV(0, vreg);
  issueK(0, 0);

  for (int ic = 0; ic < NPB / 32; ++ic) {
    const int buf = ic & 1;
    const bool more = (ic + 1 < NPB / 32);
    if (more) issueK((ic + 1) * 32, buf ^ 1);   // prefetch next K chunk
    storeV(vreg);                               // publish this chunk's V
    if (more) prefV((ic + 1) * 32, vreg);       // prefetch next V chunk
    if (more) asm volatile("s_wait_asynccnt 0x8" ::: "memory");  // cur K done
    else      asm volatile("s_wait_asynccnt 0x0" ::: "memory");
    __syncthreads();
    const u16* Kb = Kl + buf * KBUF;

    // ---- S = Q K^T : 16 rows x 32 keys, f32 accum (q pre-scaled) ----
    v8f s0 = z8, s1 = z8;
    for (int kt = 0; kt < 8; ++kt) {
      Frag16 b0, b1;  // B[k][n] = K[key=n][chan=k] -> row-major Kb works
      int co = kt * 32 + hi * 16;
      b0.q[0] = *(const uint4*)(Kb + ln * LDK + co);
      b0.q[1] = *(const uint4*)(Kb + ln * LDK + co + 8);
      b1.q[0] = *(const uint4*)(Kb + (16 + ln) * LDK + co);
      b1.q[1] = *(const uint4*)(Kb + (16 + ln) * LDK + co + 8);
      s0 = wmma_f16(qa[kt].v, b0.v, s0);
      s1 = wmma_f16(qa[kt].v, b1.v, s1);
    }

    // ---- online softmax (rows live across 16-lane halves of the wave) ----
    u16* Pw = Pl + wave * 16 * LDP;
    for (int j = 0; j < 8; ++j) {
      float mx = fmaxf(s0[j], s1[j]);
      mx = fmaxf(mx, __shfl_xor(mx, 1));
      mx = fmaxf(mx, __shfl_xor(mx, 2));
      mx = fmaxf(mx, __shfl_xor(mx, 4));
      mx = fmaxf(mx, __shfl_xor(mx, 8));
      float mn = fmaxf(m[j], mx);
      float corr = __expf(m[j] - mn);
      float p0 = __expf(s0[j] - mn);
      float p1 = __expf(s1[j] - mn);
      float rs = p0 + p1;
      rs += __shfl_xor(rs, 1);
      rs += __shfl_xor(rs, 2);
      rs += __shfl_xor(rs, 4);
      rs += __shfl_xor(rs, 8);
      l[j] = l[j] * corr + rs;
      m[j] = mn;
      for (int nt = 0; nt < 16; ++nt) o[nt][j] *= corr;
      int row = j + 8 * hi;                  // C-layout row this lane owns
      Pw[row * LDP + ln]      = f2h(p0);     // keys 0..15
      Pw[row * LDP + 16 + ln] = f2h(p1);     // keys 16..31
    }
    // same-wave LDS RAW: drain DS counter before re-reading P as an A-frag
    asm volatile("s_wait_dscnt 0" ::: "memory");
    Frag16 pa;
    pa.q[0] = *(const uint4*)(Pw + ln * LDP + hi * 16);
    pa.q[1] = *(const uint4*)(Pw + ln * LDP + hi * 16 + 8);

    // ---- O += P @ V : B[k][n] = V[key=k][chan=n] -> use transposed Vt ----
    for (int nt = 0; nt < 16; ++nt) {
      Frag16 bv;
      int ch = nt * 16 + ln;
      bv.q[0] = *(const uint4*)(Vt + ch * LDV + hi * 16);
      bv.q[1] = *(const uint4*)(Vt + ch * LDV + hi * 16 + 8);
      o[nt] = wmma_f16(pa.v, bv.v, o[nt]);
    }
    __syncthreads();   // Vt / K buffers free for reuse
  }

  // ---- finalize: O /= l, store f16 ----
  for (int j = 0; j < 8; ++j) {
    float inv = 1.f / l[j];
    int row = qbase + j + 8 * hi;
    u16* orow = O + bOff + (size_t)row * CCH;
    for (int nt = 0; nt < 16; ++nt)
      orow[nt * 16 + ln] = f2h(o[nt][j] * inv);
  }
}

// --------------------------- proj GEMM + residual --------------------------
// out_f32 = ao @ Wp + bp + xn
__global__ __launch_bounds__(128) void proj_gemm_kernel(
    const u16* __restrict__ Ah, const u16* __restrict__ Wp,
    const float* __restrict__ bp, const float* __restrict__ xn,
    float* __restrict__ out) {
  __shared__ __align__(16) u16 As[64 * LDA];
  __shared__ __align__(16) u16 Bs[64 * LDB];
  const int tid = threadIdx.x, wave = tid >> 5, lane = tid & 31;
  const int hi = lane >> 4, ln = lane & 15;
  const int mbase = blockIdx.x * 64, nbase = blockIdx.y * 64;
  const v8f z8 = {0.f, 0.f, 0.f, 0.f, 0.f, 0.f, 0.f, 0.f};
  v8f acc[4] = {z8, z8, z8, z8};

  for (int kb = 0; kb < CCH; kb += 32) {
    {
      int r = tid >> 1, cs = (tid & 1) * 16;
      const uint4* g = (const uint4*)(Ah + (size_t)(mbase + r) * CCH + kb + cs);
      *(uint4*)(As + r * LDA + cs)     = g[0];
      *(uint4*)(As + r * LDA + cs + 8) = g[1];
    }
    {
      int col = tid >> 1, ks = (tid & 1) * 16;
      __align__(16) u16 tmp[16];
      for (int j = 0; j < 16; ++j)
        tmp[j] = Wp[(size_t)(kb + ks + j) * CCH + nbase + col];
      *(uint4*)(Bs + col * LDB + ks)     = *(uint4*)tmp;
      *(uint4*)(Bs + col * LDB + ks + 8) = *(uint4*)(tmp + 8);
    }
    __syncthreads();
    Frag16 a;
    {
      int r = wave * 16 + ln, kh = hi * 16;
      a.q[0] = *(const uint4*)(As + r * LDA + kh);
      a.q[1] = *(const uint4*)(As + r * LDA + kh + 8);
    }
    for (int nt = 0; nt < 4; ++nt) {
      Frag16 b;
      int col = nt * 16 + ln, kk = hi * 16;
      b.q[0] = *(const uint4*)(Bs + col * LDB + kk);
      b.q[1] = *(const uint4*)(Bs + col * LDB + kk + 8);
      acc[nt] = wmma_f16(a.v, b.v, acc[nt]);
    }
    __syncthreads();
  }
  for (int nt = 0; nt < 4; ++nt) {
    int col = nbase + nt * 16 + ln;
    float bsv = bp[col];
    for (int j = 0; j < 8; ++j) {
      int row = mbase + wave * 16 + j + 8 * hi;
      size_t idx = (size_t)row * CCH + col;
      out[idx] = acc[nt][j] + bsv + xn[idx];
    }
  }
}

// --------------------------- launch ---------------------------------------
extern "C" void kernel_launch(void* const* d_in, const int* in_sizes, int n_in,
                              void* d_out, int out_size, void* d_ws,
                              size_t ws_size, hipStream_t stream) {
  const float* x     = (const float*)d_in[0];
  const float* gamma = (const float*)d_in[1];
  const float* beta  = (const float*)d_in[2];
  const float* Wq    = (const float*)d_in[3];
  const float* bq    = (const float*)d_in[4];
  const float* Wk    = (const float*)d_in[5];
  const float* bk    = (const float*)d_in[6];
  const float* Wv    = (const float*)d_in[7];
  const float* bv    = (const float*)d_in[8];
  const float* Wp    = (const float*)d_in[9];
  const float* bp    = (const float*)d_in[10];
  float* out = (float*)d_out;

  char* ws = (char*)d_ws;
  float* xn  = (float*)(ws + 0);
  u16*   xh  = (u16*)(ws + 16777216);
  u16*   wh  = (u16*)(ws + 25165824);   // [q|k|v|p] f16 weights
  u16*   q   = (u16*)(ws + 25690112);
  u16*   k   = (u16*)(ws + 34078720);
  u16*   v   = (u16*)(ws + 42467328);
  u16*   ao  = (u16*)(ws + 50855936);
  float* stats = (float*)(ws + 59244544);

  gn_stats_kernel<<<32, 256, 0, stream>>>(x, stats);
  gn_apply_kernel<<<4096, 256, 0, stream>>>(x, gamma, beta, stats, xn, xh);

  w2h_kernel<<<256, 256, 0, stream>>>(Wq, wh + 0 * 65536, 65536);
  w2h_kernel<<<256, 256, 0, stream>>>(Wk, wh + 1 * 65536, 65536);
  w2h_kernel<<<256, 256, 0, stream>>>(Wv, wh + 2 * 65536, 65536);
  w2h_kernel<<<256, 256, 0, stream>>>(Wp, wh + 3 * 65536, 65536);

  qkv_gemm_kernel<<<dim3(NTOK / 64, CCH / 64, 3), 128, 0, stream>>>(
      xh, wh, bq, bk, bv, q, k, v);

  flash_attn_kernel<<<dim3(NPB / 64, 4), 128, 0, stream>>>(q, k, v, ao);

  proj_gemm_kernel<<<dim3(NTOK / 64, CCH / 64), 128, 0, stream>>>(
      ao, wh + 3 * 65536, bp, xn, out);
}
